// LinearAttention_73383811219581
// MI455X (gfx1250) — matrix operand order repaired and checked by reading, not compiled, not run on hardware
//
#include <hip/hip_runtime.h>
#include <hip/hip_bf16.h>
#include <math.h>

typedef __attribute__((ext_vector_type(4)))  __bf16 v4bf;
typedef __attribute__((ext_vector_type(8)))  __bf16 v8bf;
typedef __attribute__((ext_vector_type(16))) __bf16 v16bf;
typedef __attribute__((ext_vector_type(8)))  float  v8f;

#define BT   16384   // B*T
#define TT   4096
#define CC   1024
#define HH   16
#define DD   64
#define NQKV 3072

__device__ __forceinline__ __bf16 f2bf(float f) { return (__bf16)f; }
__device__ __forceinline__ float  phi_f(float x) { return x > 0.f ? x + 1.f : __expf(x); }

// Flat LDS address -> 32-bit LDS offset (ISA 10.2: LDS offset is the low 32 bits of the address)
__device__ __forceinline__ uint32_t lds_off(const void* p) {
  return (uint32_t)(uintptr_t)p;
}

// CDNA5 async memory->LDS copy, 16 bytes, tracked by ASYNCcnt.
__device__ __forceinline__ void async_copy_b128(uint32_t loff, const void* gptr) {
  asm volatile("global_load_async_to_lds_b128 %0, %1, off"
               :: "v"(loff), "v"(gptr) : "memory");
}
__device__ __forceinline__ void wait_async_le4() {
  asm volatile("s_wait_asynccnt 0x4" ::: "memory");
}
__device__ __forceinline__ void wait_async_0() {
  asm volatile("s_wait_asynccnt 0x0" ::: "memory");
}

// Build a 16x(K=32) bf16 WMMA fragment from two 16-byte chunks.
__device__ __forceinline__ v16bf ldfrag(const __bf16* p0, const __bf16* p1) {
  v8bf lo = *(const v8bf*)p0;
  v8bf hi = *(const v8bf*)p1;
  v16bf r;
#pragma unroll
  for (int i = 0; i < 8; ++i) { r[i] = lo[i]; r[8 + i] = hi[i]; }
  return r;
}

// ---------------------------------------------------------------------------
// fp32 -> bf16 bulk convert (one pass; feeds the async-LDS GEMM path)
// ---------------------------------------------------------------------------
__global__ __launch_bounds__(256) void cvt_kernel(const float* __restrict__ src,
                                                  __bf16* __restrict__ dst, int n4) {
  int i = blockIdx.x * 256 + threadIdx.x;
  if (i < n4) {
    float4 f = ((const float4*)src)[i];
    v4bf o; o[0] = f2bf(f.x); o[1] = f2bf(f.y); o[2] = f2bf(f.z); o[3] = f2bf(f.w);
    ((v4bf*)dst)[i] = o;
  }
}

// ---------------------------------------------------------------------------
// Tiled GEMM: C[M,N] = A[M,K] * B[N,K]^T  (bf16 operands, K-contiguous)
// Block tile 128x128, BK=32, 256 threads = 8 waves (4x2), wave tile 32x64.
// Double-buffered LDS fed by global_load_async_to_lds_b128 (ASYNCcnt).
// MODE 0: qkv epilogue (phi on q/k; q row-major, k/v transposed to (b,h,d,t))
// MODE 1: proj epilogue (fp32 out + bias)
// ---------------------------------------------------------------------------
template <int MODE>
__global__ __launch_bounds__(256) void gemm_kernel(
    const __bf16* __restrict__ Ab, const __bf16* __restrict__ Bb,
    const float* __restrict__ bias,
    __bf16* __restrict__ qb, __bf16* __restrict__ kT, __bf16* __restrict__ vT,
    float* __restrict__ out, int K) {
  __shared__ __bf16 As[2][128 * 40];  // stride 40: 16B-aligned rows, conflict-free
  __shared__ __bf16 Bs[2][128 * 40];

  const int tid = threadIdx.x;
  const int lane = tid & 31;
  const int w = tid >> 5;
  const int wm = w & 3, wn = w >> 2;
  const int m0 = blockIdx.y * 128, n0 = blockIdx.x * 128;

  // Per-thread copy slots: 512 x v8bf per tile -> 2 slots each for A and B.
  const int r0 = tid >> 2, c0 = (tid & 3) * 8;            // slot tid
  const int r1 = (tid + 256) >> 2, c1 = ((tid + 256) & 3) * 8;  // slot tid+256

  auto issue = [&](int buf, int k0) {
    async_copy_b128(lds_off(&As[buf][r0 * 40 + c0]), &Ab[(size_t)(m0 + r0) * K + k0 + c0]);
    async_copy_b128(lds_off(&As[buf][r1 * 40 + c1]), &Ab[(size_t)(m0 + r1) * K + k0 + c1]);
    async_copy_b128(lds_off(&Bs[buf][r0 * 40 + c0]), &Bb[(size_t)(n0 + r0) * K + k0 + c0]);
    async_copy_b128(lds_off(&Bs[buf][r1 * 40 + c1]), &Bb[(size_t)(n0 + r1) * K + k0 + c1]);
  };

  v8f acc[2][4] = {};
  const int nk = K / 32;
  issue(0, 0);

  for (int i = 0; i < nk; i++) {
    const int buf = i & 1;
    if (i + 1 < nk) {
      issue(buf ^ 1, (i + 1) * 32);
      wait_async_le4();   // in-order completion: tile i has landed in LDS
    } else {
      wait_async_0();
    }
    __syncthreads();

    v16bf afr[2], bfr[4];
#pragma unroll
    for (int sm = 0; sm < 2; sm++) {
      int row = wm * 32 + sm * 16 + (lane & 15);
      int kb = (lane >> 4) * 8;
      afr[sm] = ldfrag(&As[buf][row * 40 + kb], &As[buf][row * 40 + kb + 16]);
    }
#pragma unroll
    for (int sn = 0; sn < 4; sn++) {
      int n = wn * 64 + sn * 16 + (lane & 15);
      int ks = (lane >> 4) * 8;
      bfr[sn] = ldfrag(&Bs[buf][n * 40 + ks], &Bs[buf][n * 40 + ks + 16]);
    }
#pragma unroll
    for (int sm = 0; sm < 2; sm++)
#pragma unroll
      for (int sn = 0; sn < 4; sn++)
        acc[sm][sn] = __builtin_amdgcn_wmma_f32_16x16x32_bf16(
            false, afr[sm], false, bfr[sn], (short)0, acc[sm][sn], false, false);
    __syncthreads();  // all waves done reading buf before it is refilled (i+2)
  }

  // ---- epilogue ----
#pragma unroll
  for (int sm = 0; sm < 2; sm++) {
#pragma unroll
    for (int sn = 0; sn < 4; sn++) {
#pragma unroll
      for (int r = 0; r < 8; r++) {
        int m = m0 + wm * 32 + sm * 16 + r + ((lane >> 4) << 3);
        int n = n0 + wn * 64 + sn * 16 + (lane & 15);
        float v = acc[sm][sn][r];
        if (MODE == 1) {
          out[(size_t)m * CC + n] = v + bias[n];
        } else {
          int b = m >> 12, t = m & (TT - 1);
          if (n < CC) {                     // q: phi, row-major
            qb[(size_t)m * CC + n] = f2bf(phi_f(v));
          } else if (n < 2 * CC) {          // k: phi, transposed (b,h,d,t)
            int e = n - CC, h = e >> 6, d = e & 63;
            kT[((size_t)((b * HH + h) * DD + d)) * TT + t] = f2bf(phi_f(v));
          } else {                          // v: transposed (b,h,m,t)
            int e = n - 2 * CC, h = e >> 6, mp = e & 63;
            vT[((size_t)((b * HH + h) * DD + mp)) * TT + t] = f2bf(v);
          }
        }
      }
    }
  }
}

// ---------------------------------------------------------------------------
// k_sum[b,h,d] = sum_t kT[b,h,d,t]
// ---------------------------------------------------------------------------
__global__ __launch_bounds__(256) void ksum_kernel(const __bf16* __restrict__ kT,
                                                   float* __restrict__ ksum) {
  __shared__ float red[256];
  int bh = blockIdx.x;
  int d = threadIdx.x & 63, part = threadIdx.x >> 6;
  const __bf16* p = kT + (size_t)bh * DD * TT + (size_t)d * TT + part * (TT / 4);
  float s = 0.f;
  for (int t = 0; t < TT / 4; t += 8) {
    v8bf v = *(const v8bf*)&p[t];
#pragma unroll
    for (int i = 0; i < 8; i++) s += (float)v[i];
  }
  red[threadIdx.x] = s;
  __syncthreads();
  if (part == 0)
    ksum[(size_t)bh * DD + d] = red[d] + red[64 + d] + red[128 + d] + red[192 + d];
}

// ---------------------------------------------------------------------------
// kv[b,h] = K^T V (64x64); 8 waves split T, reduce via ds_add_f32, store kvT bf16
// ---------------------------------------------------------------------------
__global__ __launch_bounds__(256) void kv_kernel(const __bf16* __restrict__ kT,
                                                 const __bf16* __restrict__ vT,
                                                 __bf16* __restrict__ kvT) {
  __shared__ float red[DD * DD];
  int bh = blockIdx.x;
  const __bf16* kb = kT + (size_t)bh * DD * TT;
  const __bf16* vb = vT + (size_t)bh * DD * TT;
  int tid = threadIdx.x, lane = tid & 31, w = tid >> 5;
  for (int i = tid; i < DD * DD; i += 256) red[i] = 0.f;
  __syncthreads();

  v8f acc[4][4] = {};
  for (int s = 0; s < 16; s++) {
    int t0 = w * 512 + s * 32;
    v16bf afr[4], bfr[4];
#pragma unroll
    for (int sm = 0; sm < 4; sm++) {
      int dr = sm * 16 + (lane & 15);
      int kb2 = (lane >> 4) * 8;
      const __bf16* p = kb + (size_t)dr * TT + t0 + kb2;
      afr[sm] = ldfrag(p, p + 16);
    }
#pragma unroll
    for (int sn = 0; sn < 4; sn++) {
      int mr = sn * 16 + (lane & 15);
      int ks = (lane >> 4) * 8;
      const __bf16* p = vb + (size_t)mr * TT + t0 + ks;
      bfr[sn] = ldfrag(p, p + 16);
    }
#pragma unroll
    for (int sm = 0; sm < 4; sm++)
#pragma unroll
      for (int sn = 0; sn < 4; sn++)
        acc[sm][sn] = __builtin_amdgcn_wmma_f32_16x16x32_bf16(
            false, afr[sm], false, bfr[sn], (short)0, acc[sm][sn], false, false);
  }
#pragma unroll
  for (int sm = 0; sm < 4; sm++)
#pragma unroll
    for (int sn = 0; sn < 4; sn++)
#pragma unroll
      for (int r = 0; r < 8; r++) {
        int d = sm * 16 + r + ((lane >> 4) << 3);
        int mcol = sn * 16 + (lane & 15);
        atomicAdd(&red[d * 64 + mcol], acc[sm][sn][r]);
      }
  __syncthreads();
  for (int i = tid; i < DD * DD; i += 256) {
    int d = i >> 6, mcol = i & 63;   // store transposed: kvT[m][d]
    kvT[(size_t)bh * DD * DD + mcol * DD + d] = f2bf(red[i]);
  }
}

// ---------------------------------------------------------------------------
// out[t, :] = (q[t,:] @ kv) / (q[t,:].k_sum + eps); 128 t-rows per block
// ---------------------------------------------------------------------------
__global__ __launch_bounds__(256) void attn_kernel(const __bf16* __restrict__ qb,
                                                   const __bf16* __restrict__ kvT,
                                                   const float* __restrict__ ksum,
                                                   __bf16* __restrict__ attnb) {
  __shared__ float ks_l[DD];
  __shared__ float nrm[128];
  int bh = blockIdx.x >> 5, tb = blockIdx.x & 31;
  int b = bh >> 4, h = bh & 15;
  int t0 = tb * 128;
  int tid = threadIdx.x, lane = tid & 31, w = tid >> 5;

  if (tid < DD) ks_l[tid] = ksum[(size_t)bh * DD + tid];
  __syncthreads();
  if (tid < 128) {
    const __bf16* qr = qb + (size_t)(b * TT + t0 + tid) * CC + h * DD;
    float s = 1e-6f;
    for (int d = 0; d < DD; d += 8) {
      v8bf v = *(const v8bf*)&qr[d];
#pragma unroll
      for (int i = 0; i < 8; i++) s += (float)v[i] * ks_l[d + i];
    }
    nrm[tid] = s;
  }
  __syncthreads();

  const __bf16* kvb = kvT + (size_t)bh * DD * DD;
  v8f acc[4] = {};
#pragma unroll
  for (int kk = 0; kk < 64; kk += 32) {
    int trow = w * 16 + (lane & 15);
    int kb2 = kk + (lane >> 4) * 8;
    const __bf16* qp = qb + (size_t)(b * TT + t0 + trow) * CC + h * DD + kb2;
    v16bf af = ldfrag(qp, qp + 16);
#pragma unroll
    for (int sn = 0; sn < 4; sn++) {
      int n = sn * 16 + (lane & 15);
      int ks = kk + (lane >> 4) * 8;
      v16bf bf_ = ldfrag(&kvb[n * DD + ks], &kvb[n * DD + ks + 16]);
      acc[sn] = __builtin_amdgcn_wmma_f32_16x16x32_bf16(
          false, af, false, bf_, (short)0, acc[sn], false, false);
    }
  }
#pragma unroll
  for (int sn = 0; sn < 4; sn++)
#pragma unroll
    for (int r = 0; r < 8; r++) {
      int trl = w * 16 + r + ((lane >> 4) << 3);
      int n = sn * 16 + (lane & 15);
      float v = acc[sn][r] / nrm[trl];
      attnb[(size_t)(b * TT + t0 + trl) * CC + h * DD + n] = f2bf(v);
    }
}

// ---------------------------------------------------------------------------
extern "C" void kernel_launch(void* const* d_in, const int* in_sizes, int n_in,
                              void* d_out, int out_size, void* d_ws, size_t ws_size,
                              hipStream_t stream) {
  const float* x = (const float*)d_in[0];
  const float* Wqkv = (const float*)d_in[1];
  const float* Wproj = (const float*)d_in[2];
  const float* bproj = (const float*)d_in[3];
  float* out = (float*)d_out;

  char* p = (char*)d_ws;
  const size_t big = (size_t)BT * CC * sizeof(__bf16);  // 32 MiB each
  __bf16* qb = (__bf16*)p;      p += big;
  __bf16* kT = (__bf16*)p;      p += big;
  __bf16* vT = (__bf16*)p;      p += big;
  __bf16* attnb = (__bf16*)p;   p += big;
  __bf16* xb = (__bf16*)p;      p += big;                         // bf16 copy of x
  __bf16* wqkvb = (__bf16*)p;   p += (size_t)NQKV * CC * sizeof(__bf16);
  __bf16* wprojb = (__bf16*)p;  p += (size_t)CC * CC * sizeof(__bf16);
  __bf16* kvT = (__bf16*)p;     p += (size_t)64 * DD * DD * sizeof(__bf16);
  float* ksum = (float*)p;

  // one-pass fp32 -> bf16 conversions
  cvt_kernel<<<(BT * CC / 4) / 256, 256, 0, stream>>>(x, xb, BT * CC / 4);
  cvt_kernel<<<(NQKV * CC / 4) / 256, 256, 0, stream>>>(Wqkv, wqkvb, NQKV * CC / 4);
  cvt_kernel<<<(CC * CC / 4) / 256, 256, 0, stream>>>(Wproj, wprojb, CC * CC / 4);

  gemm_kernel<0><<<dim3(NQKV / 128, BT / 128), 256, 0, stream>>>(
      xb, wqkvb, nullptr, qb, kT, vT, nullptr, CC);
  ksum_kernel<<<64, 256, 0, stream>>>(kT, ksum);
  kv_kernel<<<64, 256, 0, stream>>>(kT, vT, kvT);
  attn_kernel<<<64 * 32, 256, 0, stream>>>(qb, kvT, ksum, attnb);
  gemm_kernel<1><<<dim3(CC / 128, BT / 128), 256, 0, stream>>>(
      attnb, wprojb, bproj, nullptr, nullptr, nullptr, out, CC);
}
